// MoETransformerLayer_21655225106532
// MI455X (gfx1250) — matrix operand order repaired and checked by reading, not compiled, not run on hardware
//
#include <hip/hip_runtime.h>
#include <hip/hip_bf16.h>
#include <math.h>

// ---------------------------------------------------------------- constants
#define S_   1024
#define B_   4
#define E_   1024
#define H_   16
#define DH_  64
#define F_   4096
#define NE_  8
#define NTOK (S_ * B_)

typedef __attribute__((ext_vector_type(16))) __bf16 v16bf;
typedef __attribute__((ext_vector_type(8)))  float  v8f;

// ---------------------------------------------------------------- helpers
__device__ __forceinline__ unsigned short f32_to_bf16_bits(float f) {
    unsigned int x = __float_as_uint(f);
    unsigned int r = x + 0x7FFFu + ((x >> 16) & 1u);   // round-to-nearest-even
    return (unsigned short)(r >> 16);
}
__device__ __forceinline__ float bf16_bits_to_f32(unsigned short u) {
    return __uint_as_float(((unsigned int)u) << 16);
}

// Async global->LDS copy (CDNA5): per-lane 16B, tracked by ASYNCcnt.
// VDST operand = 32-bit LDS byte offset (low 32 bits of flat shared address).
__device__ __forceinline__ void async_copy_b128(const void* gsrc, void* ldst) {
    unsigned lds = (unsigned)(unsigned long long)ldst;
    unsigned long long ga = (unsigned long long)gsrc;
    asm volatile("global_load_async_to_lds_b128 %0, %1, off"
                 :: "v"(lds), "v"(ga) : "memory");
}

// Load 16 bf16 (two 16B chunks) from LDS into a WMMA fragment.
__device__ __forceinline__ v16bf load_frag16(const unsigned short* p, int off2) {
    union { uint4 q[2]; v16bf v; } u;
    u.q[0] = *reinterpret_cast<const uint4*>(p);
    u.q[1] = *reinterpret_cast<const uint4*>(p + off2);
    return u.v;
}

// ---------------------------------------------------------------- utility kernels
__global__ void cast_f32_bf16_kernel(const float* __restrict__ in,
                                     unsigned short* __restrict__ out, long long n) {
    long long i = (long long)blockIdx.x * blockDim.x + threadIdx.x;
    long long stride = (long long)gridDim.x * blockDim.x;
    for (; i < n; i += stride) out[i] = f32_to_bf16_bits(in[i]);
}

__global__ void copy_f32_kernel(const float* __restrict__ in,
                                float* __restrict__ out, long long n) {
    long long i = (long long)blockIdx.x * blockDim.x + threadIdx.x;
    long long stride = (long long)gridDim.x * blockDim.x;
    for (; i < n; i += stride) out[i] = in[i];
}

// Tiled transpose + cast: in (R x C f32) -> out (C x R bf16); batch = blockIdx.z
__global__ __launch_bounds__(256) void transpose_cast_kernel(
    const float* __restrict__ in, unsigned short* __restrict__ out, int R, int C) {
    __shared__ float tile[32][33];
    long long base = (long long)blockIdx.z * R * C;
    int c0 = blockIdx.x * 32, r0 = blockIdx.y * 32;
    int tx = threadIdx.x & 31, ty = threadIdx.x >> 5;   // 32 x 8
#pragma unroll
    for (int i = 0; i < 32; i += 8) {
        int r = r0 + ty + i, c = c0 + tx;
        tile[ty + i][tx] = (r < R && c < C) ? in[base + (long long)r * C + c] : 0.f;
    }
    __syncthreads();
#pragma unroll
    for (int i = 0; i < 32; i += 8) {
        int c = c0 + ty + i, r = r0 + tx;
        if (c < C && r < R)
            out[base + (long long)c * R + r] = f32_to_bf16_bits(tile[tx][ty + i]);
    }
}

// LayerNorm over last dim (E) -> bf16 tokens. One block per row.
__global__ __launch_bounds__(256) void layernorm_bf16_kernel(
    const float* __restrict__ x, const float* __restrict__ g,
    const float* __restrict__ b, unsigned short* __restrict__ out, int E) {
    int row = blockIdx.x;
    const float* xr = x + (long long)row * E;
    __shared__ float red[256];
    float s = 0.f;
    for (int i = threadIdx.x; i < E; i += 256) s += xr[i];
    red[threadIdx.x] = s; __syncthreads();
    for (int o = 128; o > 0; o >>= 1) {
        if (threadIdx.x < o) red[threadIdx.x] += red[threadIdx.x + o];
        __syncthreads();
    }
    float mu = red[0] / E; __syncthreads();
    float v = 0.f;
    for (int i = threadIdx.x; i < E; i += 256) { float d = xr[i] - mu; v += d * d; }
    red[threadIdx.x] = v; __syncthreads();
    for (int o = 128; o > 0; o >>= 1) {
        if (threadIdx.x < o) red[threadIdx.x] += red[threadIdx.x + o];
        __syncthreads();
    }
    float rstd = rsqrtf(red[0] / E + 1e-5f);
    for (int i = threadIdx.x; i < E; i += 256)
        out[(long long)row * E + i] = f32_to_bf16_bits((xr[i] - mu) * rstd * g[i] + b[i]);
}

// (S,B,3E) f32 -> q,k (B,H,S,DH) bf16 ; v transposed (B,H,DH,S) bf16
__global__ void split_qkv_kernel(const float* __restrict__ qkv,
                                 unsigned short* __restrict__ q,
                                 unsigned short* __restrict__ k,
                                 unsigned short* __restrict__ vt) {
    const long long total = (long long)NTOK * 3 * E_;
    long long stride = (long long)gridDim.x * blockDim.x;
    for (long long i = (long long)blockIdx.x * blockDim.x + threadIdx.x; i < total; i += stride) {
        int e3 = (int)(i % (3 * E_));
        long long tb = i / (3 * E_);
        int b = (int)(tb % B_);
        int s = (int)(tb / B_);
        int which = e3 / E_;
        int e = e3 % E_;
        int h = e / DH_;
        int d = e % DH_;
        unsigned short val = f32_to_bf16_bits(qkv[i]);
        if (which == 0)
            q[((long long)(b * H_ + h) * S_ + s) * DH_ + d] = val;
        else if (which == 1)
            k[((long long)(b * H_ + h) * S_ + s) * DH_ + d] = val;
        else
            vt[((long long)(b * H_ + h) * DH_ + d) * S_ + s] = val;
    }
}

// (B,H,S,DH) f32 -> (S*B, E) bf16
__global__ void merge_heads_kernel(const float* __restrict__ o,
                                   unsigned short* __restrict__ oc) {
    const long long total = (long long)NTOK * E_;
    long long stride = (long long)gridDim.x * blockDim.x;
    for (long long i = (long long)blockIdx.x * blockDim.x + threadIdx.x; i < total; i += stride) {
        int e = (int)(i % E_);
        long long tb = i / E_;
        int b = (int)(tb % B_);
        int s = (int)(tb / B_);
        int h = e / DH_;
        int d = e % DH_;
        oc[i] = f32_to_bf16_bits(o[((long long)(b * H_ + h) * S_ + s) * DH_ + d]);
    }
}

// row softmax (f32 in, bf16 out); one block per row of length C
__global__ __launch_bounds__(256) void softmax_bf16_kernel(
    const float* __restrict__ sc, unsigned short* __restrict__ at, int C) {
    long long row = blockIdx.x;
    const float* sr = sc + row * C;
    __shared__ float red[256];
    float mx = -3.4e38f;
    for (int i = threadIdx.x; i < C; i += 256) mx = fmaxf(mx, sr[i]);
    red[threadIdx.x] = mx; __syncthreads();
    for (int o = 128; o > 0; o >>= 1) {
        if (threadIdx.x < o) red[threadIdx.x] = fmaxf(red[threadIdx.x], red[threadIdx.x + o]);
        __syncthreads();
    }
    mx = red[0]; __syncthreads();
    float sum = 0.f;
    for (int i = threadIdx.x; i < C; i += 256) sum += __expf(sr[i] - mx);
    red[threadIdx.x] = sum; __syncthreads();
    for (int o = 128; o > 0; o >>= 1) {
        if (threadIdx.x < o) red[threadIdx.x] += red[threadIdx.x + o];
        __syncthreads();
    }
    float inv = 1.f / red[0];
    for (int i = threadIdx.x; i < C; i += 256)
        at[row * C + i] = f32_to_bf16_bits(__expf(sr[i] - mx) * inv);
}

// gating: logits = tok @ gate_w.T, softmax, top-2 renorm -> dense comb (N, NE)
__global__ void gate_topk_kernel(const unsigned short* __restrict__ tok,
                                 const float* __restrict__ gw,
                                 float* __restrict__ comb, int n) {
    int t = blockIdx.x * blockDim.x + threadIdx.x;
    if (t >= n) return;
    float logits[NE_];
    const unsigned short* tr = tok + (long long)t * E_;
    for (int e = 0; e < NE_; ++e) {
        const float* wr = gw + (long long)e * E_;
        float s = 0.f;
        for (int i = 0; i < E_; ++i) s += bf16_bits_to_f32(tr[i]) * wr[i];
        logits[e] = s;
    }
    float mx = logits[0];
    for (int e = 1; e < NE_; ++e) mx = fmaxf(mx, logits[e]);
    float p[NE_], sum = 0.f;
    for (int e = 0; e < NE_; ++e) { p[e] = __expf(logits[e] - mx); sum += p[e]; }
    for (int e = 0; e < NE_; ++e) p[e] /= sum;
    int i0 = 0;
    for (int e = 1; e < NE_; ++e) if (p[e] > p[i0]) i0 = e;
    int i1 = (i0 == 0) ? 1 : 0;
    for (int e = 0; e < NE_; ++e) if (e != i0 && p[e] > p[i1]) i1 = e;
    float d = p[i0] + p[i1];
    for (int e = 0; e < NE_; ++e) comb[(long long)t * NE_ + e] = 0.f;
    comb[(long long)t * NE_ + i0] = p[i0] / d;
    comb[(long long)t * NE_ + i1] = p[i1] / d;
}

// ---------------------------------------------------------------- WMMA GEMM
// C[M,N] = alpha * A[M,K] (bf16, row-major) * B[K,N] + epilogue.
// B is n-major: element (k,n) at Bm[n*ldb + k]  (all weights pre-transposed).
// Double-buffered async global->LDS pipeline (ASYNCcnt partial waits).
#define BM 128
#define BN 128
#define BKW 64
#define KPAD 8

constexpr int EPI_F32        = 0;  // Cf = v
constexpr int EPI_F32_RESID  = 1;  // Cf = v + resid
constexpr int EPI_BF16_GELU  = 2;  // Cb = bf16(gelu(v))
constexpr int EPI_ACC_SCALED = 3;  // Cf += rowscale[m*rs_stride] * v

template <int EPI>
__global__ __launch_bounds__(256) void gemm_bf16_wmma(
    const unsigned short* __restrict__ A, const unsigned short* __restrict__ Bm,
    const float* __restrict__ bias, const float* __restrict__ resid,
    const float* __restrict__ rowscale, int rs_stride,
    float* __restrict__ Cf, unsigned short* __restrict__ Cb,
    int M, int N, int K, int lda, int ldb, int ldc, float alpha,
    long long strideA, long long strideB, long long strideC) {
    int z = blockIdx.z;
    A  += (long long)z * strideA;
    Bm += (long long)z * strideB;
    long long cbase = (long long)z * strideC;

    int m0 = blockIdx.y * BM;
    int n0 = blockIdx.x * BN;
    const bool nfull = (n0 + BN) <= N;

    __shared__ unsigned short As[2][BM][BKW + KPAD];   // ping-pong A tiles
    __shared__ unsigned short Bt[2][BN][BKW + KPAD];   // ping-pong B tiles (Bt[n][k])

    int tid = threadIdx.x;
    int wave = tid >> 5;
    int lane = tid & 31;
    int lhalf = lane >> 4;   // 0 | 1
    int lmod = lane & 15;

    int wm = (wave & 3) * 32;   // wave row base (4 x 32 = 128)
    int wn = (wave >> 2) * 64;  // wave col base (2 x 64 = 128)

    v8f acc[2][4];
#pragma unroll
    for (int im = 0; im < 2; ++im)
#pragma unroll
        for (int in = 0; in < 4; ++in)
#pragma unroll
            for (int r = 0; r < 8; ++r) acc[im][in][r] = 0.f;

    int row = tid >> 1;            // 0..127 (A row / B col within tile)
    int cs = (tid & 1) * 32;       // K segment start (32 elems = 64 B)

    // issue one tile's async copies into LDS buffer `buf`
    auto stage = [&](int k0, int buf) {
        const unsigned short* srcA = A + (long long)(m0 + row) * lda + k0 + cs;
        unsigned short* dstA = &As[buf][row][cs];
#pragma unroll
        for (int i = 0; i < 4; ++i) async_copy_b128(srcA + i * 8, dstA + i * 8);
        const unsigned short* srcB = Bm + (long long)(n0 + row) * ldb + k0 + cs;
        unsigned short* dstB = &Bt[buf][row][cs];
        if (nfull) {
#pragma unroll
            for (int i = 0; i < 4; ++i) async_copy_b128(srcB + i * 8, dstB + i * 8);
        } else {
            bool ok = (n0 + row) < N;
#pragma unroll
            for (int i = 0; i < 32; ++i) dstB[i] = ok ? srcB[i] : (unsigned short)0;
        }
    };

    stage(0, 0);                   // prologue: tile 0 in flight
    int cur = 0;
    for (int k0 = 0; k0 < K; k0 += BKW) {
        const bool hasnext = (k0 + BKW) < K;
        if (hasnext) stage(k0 + BKW, cur ^ 1);   // issue-ahead into other buffer

        // Drain only the current tile's async copies (in-order completion):
        // steady state keeps the next tile's 8 (or 4) copies in flight.
        if (!hasnext)   asm volatile("s_wait_asynccnt 0x0" ::: "memory");
        else if (nfull) asm volatile("s_wait_asynccnt 0x8" ::: "memory");
        else            asm volatile("s_wait_asynccnt 0x4" ::: "memory");
        __syncthreads();

        // ---- fragments + WMMA (two k-steps of 32) on buffer `cur` ----
#pragma unroll
        for (int ks = 0; ks < 2; ++ks) {
            int kb = ks * 32;
            v16bf afrag[2];
#pragma unroll
            for (int im = 0; im < 2; ++im)
                afrag[im] = load_frag16(&As[cur][wm + im * 16 + lmod][kb + lhalf * 8], 16);
#pragma unroll
            for (int in = 0; in < 4; ++in) {
                v16bf bfrag = load_frag16(&Bt[cur][wn + in * 16 + lmod][kb + lhalf * 8], 16);
#pragma unroll
                for (int im = 0; im < 2; ++im)
                    acc[im][in] = __builtin_amdgcn_wmma_f32_16x16x32_bf16(
                        false, afrag[im], false, bfrag, (short)0, acc[im][in], false, false);
            }
        }
        __syncthreads();           // everyone done reading `cur` before it is restaged
        cur ^= 1;
    }

    // ---- epilogue ----  C layout: row = r + 8*lhalf, col = lmod
#pragma unroll
    for (int im = 0; im < 2; ++im) {
#pragma unroll
        for (int in = 0; in < 4; ++in) {
            int n = n0 + wn + in * 16 + lmod;
            if (n >= N) continue;
            float bv = bias ? bias[n] : 0.f;
#pragma unroll
            for (int r = 0; r < 8; ++r) {
                int m = m0 + wm + im * 16 + r + lhalf * 8;
                long long ci = cbase + (long long)m * ldc + n;
                float v = acc[im][in][r] * alpha + bv;
                if (EPI == EPI_F32) {
                    Cf[ci] = v;
                } else if (EPI == EPI_F32_RESID) {
                    Cf[ci] = v + resid[ci];
                } else if (EPI == EPI_BF16_GELU) {
                    float gv = 0.5f * v * (1.f + erff(v * 0.7071067811865476f));
                    Cb[ci] = f32_to_bf16_bits(gv);
                } else {  // EPI_ACC_SCALED
                    Cf[ci] += rowscale[(long long)m * rs_stride] * v;
                }
            }
        }
    }
}

// ---------------------------------------------------------------- launch
extern "C" void kernel_launch(void* const* d_in, const int* in_sizes, int n_in,
                              void* d_out, int out_size, void* d_ws, size_t ws_size,
                              hipStream_t stream) {
    const float* x          = (const float*)d_in[0];
    const float* ln1_g      = (const float*)d_in[1];
    const float* ln1_b      = (const float*)d_in[2];
    const float* ln2_g      = (const float*)d_in[3];
    const float* ln2_b      = (const float*)d_in[4];
    const float* in_proj_w  = (const float*)d_in[5];
    const float* in_proj_b  = (const float*)d_in[6];
    const float* out_proj_w = (const float*)d_in[7];
    const float* out_proj_b = (const float*)d_in[8];
    const float* gate_w     = (const float*)d_in[9];
    const float* w1         = (const float*)d_in[10];
    const float* b1         = (const float*)d_in[11];
    const float* w2         = (const float*)d_in[12];
    const float* b2         = (const float*)d_in[13];
    float* out = (float*)d_out;

    char* ws = (char*)d_ws;
    size_t off = 0;
    auto alloc = [&](size_t bytes) -> void* {
        void* p = (void*)(ws + off);
        off += (bytes + 255) & ~(size_t)255;
        return p;
    };

    unsigned short* wb_in  = (unsigned short*)alloc((size_t)3 * E_ * E_ * 2);
    unsigned short* wb_out = (unsigned short*)alloc((size_t)E_ * E_ * 2);
    unsigned short* w1t    = (unsigned short*)alloc((size_t)NE_ * E_ * F_ * 2); // (F,E) per e
    unsigned short* w2t    = (unsigned short*)alloc((size_t)NE_ * F_ * E_ * 2); // (E,F) per e
    unsigned short* h1     = (unsigned short*)alloc((size_t)NTOK * E_ * 2);
    float*          qkv    = (float*)alloc((size_t)NTOK * 3 * E_ * 4);
    unsigned short* qb     = (unsigned short*)alloc((size_t)NTOK * E_ * 2);
    unsigned short* kb     = (unsigned short*)alloc((size_t)NTOK * E_ * 2);
    unsigned short* vt     = (unsigned short*)alloc((size_t)NTOK * E_ * 2);   // (B,H,DH,S)
    float*          scores = (float*)alloc((size_t)B_ * H_ * S_ * S_ * 4);
    unsigned short* attn   = (unsigned short*)alloc((size_t)B_ * H_ * S_ * S_ * 2);
    float*          oheads = (float*)alloc((size_t)NTOK * E_ * 4);
    unsigned short* oc     = (unsigned short*)alloc((size_t)NTOK * E_ * 2);
    float*          x2     = (float*)alloc((size_t)NTOK * E_ * 4);
    unsigned short* h2     = (unsigned short*)alloc((size_t)NTOK * E_ * 2);
    float*          comb   = (float*)alloc((size_t)NTOK * NE_ * 4);
    unsigned short* hexp   = (unsigned short*)alloc((size_t)NTOK * F_ * 2);
    (void)ws_size; (void)in_sizes; (void)n_in; (void)out_size;

    const dim3 gsBig(4096), tb(256);

    // 0) weights -> bf16 (projections as-is = n-major; experts transposed)
    cast_f32_bf16_kernel<<<gsBig, tb, 0, stream>>>(in_proj_w, wb_in, (long long)3 * E_ * E_);
    cast_f32_bf16_kernel<<<gsBig, tb, 0, stream>>>(out_proj_w, wb_out, (long long)E_ * E_);
    {
        dim3 g1(F_ / 32, E_ / 32, NE_);   // w1 (E x F) -> w1t (F x E)
        transpose_cast_kernel<<<g1, tb, 0, stream>>>(w1, w1t, E_, F_);
        dim3 g2(E_ / 32, F_ / 32, NE_);   // w2 (F x E) -> w2t (E x F)
        transpose_cast_kernel<<<g2, tb, 0, stream>>>(w2, w2t, F_, E_);
    }

    // 1) LN1
    layernorm_bf16_kernel<<<NTOK, tb, 0, stream>>>(x, ln1_g, ln1_b, h1, E_);

    // 2) QKV = h1 @ in_proj_w.T + b   (M=4096, N=3072, K=1024)
    {
        dim3 g((3 * E_) / BN, NTOK / BM, 1);
        gemm_bf16_wmma<EPI_F32><<<g, tb, 0, stream>>>(
            h1, wb_in, in_proj_b, nullptr, nullptr, 0, qkv, nullptr,
            NTOK, 3 * E_, E_, E_, E_, 3 * E_, 1.0f, 0, 0, 0);
    }
    split_qkv_kernel<<<gsBig, tb, 0, stream>>>(qkv, qb, kb, vt);

    // 3) scores = q @ k^T / sqrt(DH)  (batched B*H; M=N=S, K=DH)
    {
        dim3 g(S_ / BN, S_ / BM, B_ * H_);
        gemm_bf16_wmma<EPI_F32><<<g, tb, 0, stream>>>(
            qb, kb, nullptr, nullptr, nullptr, 0, scores, nullptr,
            S_, S_, DH_, DH_, DH_, S_, 0.125f,
            (long long)S_ * DH_, (long long)S_ * DH_, (long long)S_ * S_);
    }
    softmax_bf16_kernel<<<B_ * H_ * S_, tb, 0, stream>>>(scores, attn, S_);

    // 4) o = attn @ v   (batched; M=S, N=DH, K=S); B = vt (DH x S per head), n-major
    {
        dim3 g((DH_ + BN - 1) / BN, S_ / BM, B_ * H_);
        gemm_bf16_wmma<EPI_F32><<<g, tb, 0, stream>>>(
            attn, vt, nullptr, nullptr, nullptr, 0, oheads, nullptr,
            S_, DH_, S_, S_, S_, DH_, 1.0f,
            (long long)S_ * S_, (long long)DH_ * S_, (long long)S_ * DH_);
    }
    merge_heads_kernel<<<gsBig, tb, 0, stream>>>(oheads, oc);

    // 5) x2 = oc @ out_proj_w.T + b + x   (M=4096, N=1024, K=1024)
    {
        dim3 g(E_ / BN, NTOK / BM, 1);
        gemm_bf16_wmma<EPI_F32_RESID><<<g, tb, 0, stream>>>(
            oc, wb_out, out_proj_b, x, nullptr, 0, x2, nullptr,
            NTOK, E_, E_, E_, E_, E_, 1.0f, 0, 0, 0);
    }

    // 6) LN2, gating, residual init of output
    layernorm_bf16_kernel<<<NTOK, tb, 0, stream>>>(x2, ln2_g, ln2_b, h2, E_);
    gate_topk_kernel<<<(NTOK + 255) / 256, tb, 0, stream>>>(h2, gate_w, comb, NTOK);
    copy_f32_kernel<<<gsBig, tb, 0, stream>>>(x2, out, (long long)NTOK * E_);

    // 7) experts (dense, matching reference): out += comb[:,e] * (gelu(h2@w1[e]+b1)@w2[e]+b2)
    for (int e = 0; e < NE_; ++e) {
        dim3 g1(F_ / BN, NTOK / BM, 1);
        gemm_bf16_wmma<EPI_BF16_GELU><<<g1, tb, 0, stream>>>(
            h2, w1t + (size_t)e * F_ * E_, b1 + (size_t)e * F_, nullptr, nullptr, 0,
            nullptr, hexp, NTOK, F_, E_, E_, E_, F_, 1.0f, 0, 0, 0);
        dim3 g2(E_ / BN, NTOK / BM, 1);
        gemm_bf16_wmma<EPI_ACC_SCALED><<<g2, tb, 0, stream>>>(
            hexp, w2t + (size_t)e * E_ * F_, b2 + (size_t)e * E_, nullptr,
            comb + e, NE_, out, nullptr, NTOK, E_, F_, F_, F_, E_, 1.0f, 0, 0, 0);
    }
}
